// ObjectEncoder_7756710937067
// MI455X (gfx1250) — matrix operand order repaired and checked by reading, not compiled, not run on hardware
//
#include <hip/hip_runtime.h>

typedef __attribute__((ext_vector_type(16))) _Float16 v16h;
typedef __attribute__((ext_vector_type(8)))  _Float16 v8h;
typedef __attribute__((ext_vector_type(8)))  float    v8f;

#define PATCH     16
#define IN_DIM    768
#define HID       384
#define NB_FRAMES 12
#define OBJ       10
#define BS_T      48
#define NTHREADS  384   // 12 waves of 32
#define NWAVES    12
#define MROWS     256   // 16x16 ROI cells
#define MC_ROWS   64    // M-chunk staged in LDS per pass

#define A_BYTES (MC_ROWS * IN_DIM * 2)   // 98304  : f16 ROI chunk (64 x 768)
#define H_BYTES (MROWS * HID * 2)        // 196608 : f16 hidden (256 x 384)

// packed-weight workspace: fragment-native f16 layout, [nt][kt][lane][16]
#define W1_ELEMS (IN_DIM * HID)          // 294912
#define W2_ELEMS (HID * IN_DIM)          // 294912
#define PACK_BYTES ((size_t)(W1_ELEMS + W2_ELEMS) * 2)   // ~1.1 MB
#define W1_NKT (IN_DIM / 32)             // 24 k-tiles
#define W2_NKT (HID / 32)                // 12 k-tiles

struct Aux {
  float wy0[PATCH], wy1[PATCH], wx0[PATCH], wx1[PATCH];
  int   yi0[PATCH], yi1[PATCH], xi0[PATCH], xi1[PATCH];
  float be[HID];        // relu(boxes/224 @ C1)
  float bemb[IN_DIM];   // relu(be @ C2)
};

#define SMEM_BYTES (A_BYTES + H_BYTES + (int)sizeof(Aux))  // ~293 KB (<= 320 KB/WGP)

// A-fragment (16x32 f16, M x K): lane l holds row M=l%16; lanes>=16 take K 16..31 (contiguous).
__device__ __forceinline__ v16h load_afrag(const _Float16* __restrict__ row, int kbase, int hi) {
  const _Float16* p = row + kbase + 16 * hi;
  v8h lo = *(const v8h*)p;          // K = kbase+16*hi .. +7
  v8h hh = *(const v8h*)(p + 8);    // K = kbase+16*hi+8 .. +15
  v16h r;
#pragma unroll
  for (int t = 0; t < 8; ++t) { r[t] = lo[t]; r[t + 8] = hh[t]; }
  return r;
}

// B-fragment (32x16 f16, K x N) from a row-major f32 matrix with leading dim ld (fallback path).
__device__ __forceinline__ v16h load_bfrag_f32(const float* __restrict__ W, int ld,
                                               int kbase, int nb, int l16, int hi) {
  const float* p = W + (size_t)(kbase + 16 * hi) * ld + nb + l16;
  v16h r;
#pragma unroll
  for (int t = 0; t < 16; ++t) r[t] = (_Float16)p[(size_t)t * ld];
  return r;
}

// B-fragment from pre-packed f16 workspace: one contiguous 32B load per lane.
__device__ __forceinline__ v16h load_bfrag_packed(const _Float16* __restrict__ Wp,
                                                  int nkt, int nt, int kt, int lane) {
  return *(const v16h*)(Wp + (((size_t)nt * nkt + kt) * 32 + lane) * 16);
}

// Repack a row-major f32 K x N weight into fragment-native f16 layout.
// dest flat index = ((nt*NKT + kt)*32 + lane)*16 + t  ->  source (k,n):
//   k = kt*32 + 16*(lane>>4) + t ; n = nt*16 + (lane&15)
__global__ void pack_weights_kernel(const float* __restrict__ W, _Float16* __restrict__ Wp,
                                    int K, int N) {
  int idx = blockIdx.x * blockDim.x + threadIdx.x;
  if (idx >= K * N) return;
  int t    = idx & 15;
  int lane = (idx >> 4) & 31;
  int frag = idx >> 9;
  int nkt = K / 32;
  int nt = frag / nkt, kt = frag % nkt;
  int k = kt * 32 + 16 * (lane >> 4) + t;
  int n = nt * 16 + (lane & 15);
  Wp[idx] = (_Float16)W[(size_t)k * N + n];
}

template <bool PACKED>
__global__ void __launch_bounds__(NTHREADS, 1)
object_encoder_kernel(const float* __restrict__ features,   // (257, 48, 768)
                      const float* __restrict__ boxes,      // (48, 10, 4)
                      const float* __restrict__ W1,         // (768, 384) f32
                      const float* __restrict__ W2,         // (384, 768) f32
                      const _Float16* __restrict__ Wp1,     // packed W1 (or null)
                      const _Float16* __restrict__ Wp2,     // packed W2 (or null)
                      const float* __restrict__ C1,         // (4, 384)
                      const float* __restrict__ C2,         // (384, 768)
                      const float* __restrict__ bcat,       // (12, 10, 768)
                      float* __restrict__ out)              // 10*48*768 tokens + 48*10 mask
{
  extern __shared__ __align__(16) char smem[];
  _Float16* A_buf = (_Float16*)smem;                 // 64 x 768 f16
  _Float16* H_buf = (_Float16*)(smem + A_BYTES);     // 256 x 384 f16
  Aux* aux = (Aux*)(smem + A_BYTES + H_BYTES);

  const int tid  = threadIdx.x;
  const int wave = tid >> 5;
  const int lane = tid & 31;
  const int l16  = lane & 15;
  const int hi   = lane >> 4;

  const int wg = blockIdx.x;       // 480 = 48 * 10
  const int o  = wg % OBJ;
  const int bt = wg / OBJ;

  const float* bx = boxes + ((size_t)bt * OBJ + o) * 4;
  const float scale = (float)PATCH / 224.0f;

  // ---- bilinear axis setup (threads 0..31) ----
  if (tid < 2 * PATCH) {
    int axis = tid >> 4;           // 0 = y, 1 = x
    int i = tid & 15;
    float c1 = (axis ? bx[0] : bx[1]) * scale - 0.5f;
    float c2 = (axis ? bx[2] : bx[3]) * scale - 0.5f;
    float bin = (c2 - c1) * (1.0f / (float)PATCH);
    float coord = c1 + ((float)i + 0.5f) * bin;
    float valid = (coord >= -1.0f && coord <= (float)PATCH) ? 1.0f : 0.0f;
    float c = fmaxf(coord, 0.0f);
    int i0 = (int)floorf(c); if (i0 > PATCH - 1) i0 = PATCH - 1;
    int i1 = i0 + 1;         if (i1 > PATCH - 1) i1 = PATCH - 1;
    float l = c - (float)i0; l = fminf(fmaxf(l, 0.0f), 1.0f);
    if (axis == 0) { aux->yi0[i] = i0; aux->yi1[i] = i1; aux->wy0[i] = (1.0f - l) * valid; aux->wy1[i] = l * valid; }
    else           { aux->xi0[i] = i0; aux->xi1[i] = i1; aux->wx0[i] = (1.0f - l) * valid; aux->wx1[i] = l * valid; }
  }

  // ---- box-embedding layer 1: be[k] = relu(sum_c (bx[c]/224) * C1[c][k]) ----
  {
    float b0 = bx[0] * (1.0f / 224.0f), b1 = bx[1] * (1.0f / 224.0f);
    float b2 = bx[2] * (1.0f / 224.0f), b3 = bx[3] * (1.0f / 224.0f);
    float s = b0 * C1[0 * HID + tid] + b1 * C1[1 * HID + tid] +
              b2 * C1[2 * HID + tid] + b3 * C1[3 * HID + tid];
    aux->be[tid] = fmaxf(s, 0.0f);
  }
  __syncthreads();

  // ---- box-embedding layer 2: bemb[d] = relu(be @ C2) ----
#pragma unroll
  for (int r = 0; r < 2; ++r) {
    int d = tid + r * NTHREADS;
    float s = 0.0f;
    for (int k = 0; k < HID; ++k) s += aux->be[k] * C2[(size_t)k * IN_DIM + d];
    aux->bemb[d] = fmaxf(s, 0.0f);
  }

  // ================= GEMM1: relu(ROI(256x768) @ W1(768x384)) -> H_buf =================
  for (int mc = 0; mc < 4; ++mc) {
    __syncthreads();   // also orders aux writes / previous-iter A_buf reads
    {
      // stage 64 ROI rows (full K=768) as f16 into A_buf; 6 threads per row
      int lr = tid / 6;
      int d0 = (tid % 6) * 128;
      int gr = mc * MC_ROWS + lr;
      int p = gr >> 4, q = gr & 15;
      const float* f00 = features + ((size_t)(1 + aux->yi0[p] * PATCH + aux->xi0[q]) * BS_T + bt) * IN_DIM;
      const float* f01 = features + ((size_t)(1 + aux->yi0[p] * PATCH + aux->xi1[q]) * BS_T + bt) * IN_DIM;
      const float* f10 = features + ((size_t)(1 + aux->yi1[p] * PATCH + aux->xi0[q]) * BS_T + bt) * IN_DIM;
      const float* f11 = features + ((size_t)(1 + aux->yi1[p] * PATCH + aux->xi1[q]) * BS_T + bt) * IN_DIM;
      float w00 = aux->wy0[p] * aux->wx0[q], w01 = aux->wy0[p] * aux->wx1[q];
      float w10 = aux->wy1[p] * aux->wx0[q], w11 = aux->wy1[p] * aux->wx1[q];
      _Float16* dst = A_buf + (size_t)lr * IN_DIM;
      for (int d = d0; d < d0 + 128; d += 4) {
        float4 a = *(const float4*)(f00 + d);
        float4 b = *(const float4*)(f01 + d);
        float4 c = *(const float4*)(f10 + d);
        float4 e = *(const float4*)(f11 + d);
        dst[d + 0] = (_Float16)(w00 * a.x + w01 * b.x + w10 * c.x + w11 * e.x);
        dst[d + 1] = (_Float16)(w00 * a.y + w01 * b.y + w10 * c.y + w11 * e.y);
        dst[d + 2] = (_Float16)(w00 * a.z + w01 * b.z + w10 * c.z + w11 * e.z);
        dst[d + 3] = (_Float16)(w00 * a.w + w01 * b.w + w10 * c.w + w11 * e.w);
      }
    }
    __syncthreads();

    // each wave owns n-tiles {2*wave, 2*wave+1}; 4 M-tiles per chunk
#pragma unroll
    for (int j = 0; j < 2; ++j) {
      int nt = 2 * wave + j;
      int nb = nt * 16;
      v8f acc[4] = {};
      for (int kk = 0; kk < IN_DIM; kk += 32) {
        v16h bfrag = PACKED ? load_bfrag_packed(Wp1, W1_NKT, nt, kk >> 5, lane)
                            : load_bfrag_f32(W1, HID, kk, nb, l16, hi);
#pragma unroll
        for (int m = 0; m < 4; ++m) {
          v16h afrag = load_afrag(A_buf + (size_t)(m * 16 + l16) * IN_DIM, kk, hi);
          acc[m] = __builtin_amdgcn_wmma_f32_16x16x32_f16(
              false, afrag, false, bfrag, (short)0, acc[m], false, false);
        }
      }
#pragma unroll
      for (int m = 0; m < 4; ++m)
#pragma unroll
        for (int v = 0; v < 8; ++v) {
          int row = mc * MC_ROWS + m * 16 + 8 * hi + v;
          H_buf[(size_t)row * HID + nb + l16] = (_Float16)fmaxf(acc[m][v], 0.0f);
        }
    }
  }
  __syncthreads();

  // ============ GEMM2: max_rows(relu(H(256x384) @ W2(384x768))) + epilogue ============
  const int frame = bt % NB_FRAMES;
#pragma unroll
  for (int j = 0; j < 4; ++j) {
    int nt = 4 * wave + j;
    int nb = nt * 16;
    v16h bfr[12];
#pragma unroll
    for (int kt = 0; kt < 12; ++kt)
      bfr[kt] = PACKED ? load_bfrag_packed(Wp2, W2_NKT, nt, kt, lane)
                       : load_bfrag_f32(W2, IN_DIM, kt * 32, nb, l16, hi);

    float cmax = 0.0f;   // relu output >= 0, so 0 is the identity for the max
    for (int m = 0; m < 16; ++m) {
      v8f acc = {};
#pragma unroll
      for (int kt = 0; kt < 12; ++kt) {
        v16h afrag = load_afrag(H_buf + (size_t)(m * 16 + l16) * HID, kt * 32, hi);
        acc = __builtin_amdgcn_wmma_f32_16x16x32_f16(
            false, afrag, false, bfr[kt], (short)0, acc, false, false);
      }
#pragma unroll
      for (int v = 0; v < 8; ++v) cmax = fmaxf(cmax, fmaxf(acc[v], 0.0f));
    }
    cmax = fmaxf(cmax, __shfl_xor(cmax, 16, 32));   // merge row-halves (wave32)
    if (hi == 0) {
      int n = nb + l16;
      float val = cmax + bcat[((size_t)frame * OBJ + o) * IN_DIM + n] + aux->bemb[n];
      out[((size_t)o * BS_T + bt) * IN_DIM + n] = val;   // (O, bs_t, d) layout
    }
  }

  if (tid == 0)
    out[(size_t)OBJ * BS_T * IN_DIM + (size_t)bt * OBJ + o] = (bx[0] == -1.0f) ? 1.0f : 0.0f;
}

extern "C" void kernel_launch(void* const* d_in, const int* in_sizes, int n_in,
                              void* d_out, int out_size, void* d_ws, size_t ws_size,
                              hipStream_t stream) {
  (void)in_sizes; (void)n_in; (void)out_size;
  const float* features = (const float*)d_in[0];
  const float* boxes    = (const float*)d_in[1];
  const float* W1       = (const float*)d_in[2];
  const float* W2       = (const float*)d_in[3];
  const float* C1       = (const float*)d_in[4];
  const float* C2       = (const float*)d_in[5];
  const float* bcat     = (const float*)d_in[6];
  float* out = (float*)d_out;

  const bool packed = (d_ws != nullptr) && (ws_size >= PACK_BYTES);
  _Float16* Wp1 = (_Float16*)d_ws;
  _Float16* Wp2 = Wp1 + W1_ELEMS;

  dim3 grid(BS_T * OBJ);     // 480 workgroups, one per (frame, object) ROI
  dim3 block(NTHREADS);      // 12 wave32 waves

  if (packed) {
    // repack weights into fragment-native f16 layout (stream-ordered, runs first)
    int threads = 256;
    hipLaunchKernelGGL(pack_weights_kernel, dim3((W1_ELEMS + threads - 1) / threads),
                       dim3(threads), 0, stream, W1, Wp1, IN_DIM, HID);
    hipLaunchKernelGGL(pack_weights_kernel, dim3((W2_ELEMS + threads - 1) / threads),
                       dim3(threads), 0, stream, W2, Wp2, HID, IN_DIM);

    (void)hipFuncSetAttribute((const void*)object_encoder_kernel<true>,
                              hipFuncAttributeMaxDynamicSharedMemorySize, SMEM_BYTES);
    hipLaunchKernelGGL(object_encoder_kernel<true>, grid, block, SMEM_BYTES, stream,
                       features, boxes, W1, W2, Wp1, Wp2, C1, C2, bcat, out);
  } else {
    (void)hipFuncSetAttribute((const void*)object_encoder_kernel<false>,
                              hipFuncAttributeMaxDynamicSharedMemorySize, SMEM_BYTES);
    hipLaunchKernelGGL(object_encoder_kernel<false>, grid, block, SMEM_BYTES, stream,
                       features, boxes, W1, W2, (const _Float16*)nullptr,
                       (const _Float16*)nullptr, C1, C2, bcat, out);
  }
}
